// frequency_selection_46145128628231
// MI455X (gfx1250) — compile-verified, matrix operand order checked by reading, not run on hardware
//
#include <hip/hip_runtime.h>
#include <math.h>

// ---------------------------------------------------------------------------
// FFT-mix-iFFT ("frequency selection") fully on CDNA5 WMMA f16 tensor path.
//   x:(16,32,256,256) f32 -> window 64x64 -> rfft2 -> complex 32->32 mix ->
//   exact GELU(Re)/GELU(Im) -> complex 32->32 mix -> irfft2 -> output f32
// All linear transforms (row rDFT, column DFT/IDFT, channel mixing) are
// GEMMs on v_wmma_f32_16x16x32_f16 (f32 accumulate). fp32 staging between
// kernels; f16 only as WMMA operands. B matrices stored TRANSPOSED [N][K]
// so both A and B fragments load as 2x contiguous 16B runs per lane.
// Each wave computes a 2x2 quad of 16x16 tiles (4 accumulators): 4 WMMAs per
// k-step share 2 A + 2 B fragments, and dependent WMMAs are 3 independent
// WMMAs apart -> the f16 WMMA 1+4 coexec hazard window is filled, no v_nops.
// ---------------------------------------------------------------------------

typedef __attribute__((ext_vector_type(16))) _Float16 v16h;
typedef __attribute__((ext_vector_type(8)))  float    v8f;

#define NWIN 256              // 16 imgs * 4 * 4 windows
#define NK   33               // rfft bins along W
// scratch: Y[win][k][c][2h+p]  (f32)  -- contiguous 32x128 chunk per (win,k)
#define Y_FLOATS (256u*33u*32u*128u)   // 34,603,008 floats
#define Y_BYTES  (Y_FLOATS*4u)         // 138,412,032 bytes (16B aligned)
// f16 matrices appended after Y (offsets in halves; all 16B aligned)
// All stored TRANSPOSED: BT[N][K]
#define FROW_OFF   0        // 96 x 64   row rDFT^T   (rows 66..95 zero)
#define FCOL_OFF   6144     // 128 x 128 column DFT^T (real-ified, interleaved)
#define W1_OFF     22528    // 64 x 64   [[W1r,W1i],[-W1i,W1r]]^T
#define W2_OFF     26624    // 64 x 64
#define FICOL_OFF  30720    // 128 x 128 column IDFT^T (unscaled)
#define FIROW_OFF  47104    // 64 x 96   Hermitian-folded inverse row^T (cols 66..95 zero)
#define MATS_TOTAL 53248

// ---- WMMA fragment helpers (wave32, 16x16x32 f16) -------------------------
// A 16x32: half t of v16h, lane L:  row = L&15,  K = (t>>3)*16 + (L&16?8:0) + (t&7)
// B loaded from B^T with the same pattern (lane carries N instead of M).
// C/D 16x16 f32: vgpr i, lane L: row = i + (L&16?8:0), col = L&15
static __device__ __forceinline__ v16h load_frag_rowK(const _Float16* base, int ld,
                                                      int row0, int k0, int lane) {
  const int r  = row0 + (lane & 15);
  const int kh = (lane & 16) ? 8 : 0;
  v16h a;
#pragma unroll
  for (int t = 0; t < 16; ++t)
    a[t] = base[r * ld + k0 + ((t >> 3) << 4) + kh + (t & 7)];
  return a;
}
static __device__ __forceinline__ void store_frag_C(float* base, int ld,
                                                    int row0, int col0, int lane, v8f acc) {
  const int c  = col0 + (lane & 15);
  const int r0 = row0 + ((lane & 16) ? 8 : 0);
#pragma unroll
  for (int i = 0; i < 8; ++i) base[(r0 + i) * ld + c] = acc[i];
}

// GEMM: C(MxN,f32) = A(MxK,f16,ld lda) * B(KxN) given as BT(NxK,f16,ld ldbT)
// M,N multiples of 32; K multiple of 32. 4 waves per block; 2x2 tile quads
// round-robined across waves; 4 independent accumulators per quad.
static __device__ __forceinline__ void gemm_f16(const _Float16* A, int lda,
                                                const _Float16* BT, int ldbT,
                                                float* C, int ldc,
                                                int M, int N, int K,
                                                int wave, int lane) {
  const int nq = N >> 5, kt = K >> 5, quads = (M >> 5) * nq;
  for (int q = wave; q < quads; q += 4) {
    const int qm = (q / nq) << 5, qn = (q % nq) << 5;
    v8f acc00 = {}, acc01 = {}, acc10 = {}, acc11 = {};
    for (int ks = 0; ks < kt; ++ks) {
      const int k0 = ks << 5;
      v16h a0 = load_frag_rowK(A,  lda,  qm,      k0, lane);
      v16h a1 = load_frag_rowK(A,  lda,  qm + 16, k0, lane);
      v16h b0 = load_frag_rowK(BT, ldbT, qn,      k0, lane);
      v16h b1 = load_frag_rowK(BT, ldbT, qn + 16, k0, lane);
      acc00 = __builtin_amdgcn_wmma_f32_16x16x32_f16(false, a0, false, b0,
                                                     (short)0, acc00, false, false);
      acc01 = __builtin_amdgcn_wmma_f32_16x16x32_f16(false, a0, false, b1,
                                                     (short)0, acc01, false, false);
      acc10 = __builtin_amdgcn_wmma_f32_16x16x32_f16(false, a1, false, b0,
                                                     (short)0, acc10, false, false);
      acc11 = __builtin_amdgcn_wmma_f32_16x16x32_f16(false, a1, false, b1,
                                                     (short)0, acc11, false, false);
    }
    store_frag_C(C, ldc, qm,      qn,      lane, acc00);
    store_frag_C(C, ldc, qm,      qn + 16, lane, acc01);
    store_frag_C(C, ldc, qm + 16, qn,      lane, acc10);
    store_frag_C(C, ldc, qm + 16, qn + 16, lane, acc11);
  }
}

// ---- K0: build f16 transform matrices (transposed) in scratch -------------
__global__ void fsel_k0_build(const float* __restrict__ w1r, const float* __restrict__ w1i,
                              const float* __restrict__ w2r, const float* __restrict__ w2i,
                              _Float16* __restrict__ mats) {
  const float STEP = 0.09817477042468103f;  // 2*pi/64
  for (int i = blockIdx.x * blockDim.x + threadIdx.x; i < MATS_TOTAL;
       i += gridDim.x * blockDim.x) {
    float v = 0.0f;
    if (i < FCOL_OFF) {                         // FROW^T [n=0..95][w=0..63]
      int n = i >> 6, w = i & 63;               // logical Frow[w][n]
      if (n < 66) {
        float t = (float)((w * (n >> 1)) & 63) * STEP;
        v = (n & 1) ? -sinf(t) : cosf(t);
      }
    } else if (i < W1_OFF) {                    // FCOL^T [n][r], forward e^{-i t}
      int i2 = i - FCOL_OFF, n = i2 >> 7, r = i2 & 127;
      int pin = r & 1, po = n & 1;
      float t = (float)(((r >> 1) * (n >> 1)) & 63) * STEP;
      v = (pin == po) ? cosf(t) : (pin ? sinf(t) : -sinf(t));
    } else if (i < W2_OFF) {                    // W1~^T [n][j]
      int i3 = i - W1_OFF, n = i3 >> 6, j = i3 & 63;
      const float* base = (j < 32) ? ((n < 32) ? w1r : w1i)
                                   : ((n < 32) ? w1i : w1r);
      float s = (j >= 32 && n < 32) ? -1.0f : 1.0f;
      v = s * base[(j & 31) * 32 + (n & 31)];
    } else if (i < FICOL_OFF) {                 // W2~^T [n][j]
      int i3 = i - W2_OFF, n = i3 >> 6, j = i3 & 63;
      const float* base = (j < 32) ? ((n < 32) ? w2r : w2i)
                                   : ((n < 32) ? w2i : w2r);
      float s = (j >= 32 && n < 32) ? -1.0f : 1.0f;
      v = s * base[(j & 31) * 32 + (n & 31)];
    } else if (i < FIROW_OFF) {                 // FICOL^T [n][r], inverse e^{+i t}
      int i2 = i - FICOL_OFF, n = i2 >> 7, r = i2 & 127;
      int pin = r & 1, po = n & 1;
      float t = (float)(((r >> 1) * (n >> 1)) & 63) * STEP;
      v = (pin == po) ? cosf(t) : (pin ? -sinf(t) : sinf(t));
    } else {                                    // FIROW^T [w=0..63][r=0..95]
      int i2 = i - FIROW_OFF, w = i2 / 96, r = i2 % 96;
      if (r < 66) {
        int k = r >> 1;
        float g = (k == 0 || k == 32) ? 1.0f : 2.0f;
        float t = (float)((w * k) & 63) * STEP;
        v = (r & 1) ? -g * sinf(t) : g * cosf(t);
      }
    }
    mats[i] = (_Float16)v;
  }
}

// ---- K1: per (win,c): row rDFT via WMMA, scatter to Y[win][k][c][2h+p] ----
__global__ __launch_bounds__(128) void fsel_k1_rowfft(const float* __restrict__ x,
                                                      float* __restrict__ Y,
                                                      const _Float16* __restrict__ mats) {
  __shared__ _Float16 sX[64 * 72];
  __shared__ float    sO[64 * 97];
  const int tid = threadIdx.x, wave = tid >> 5, lane = tid & 31;
  const int win = blockIdx.x >> 5, c = blockIdx.x & 31;
  const int b = win >> 4, wh = (win >> 2) & 3, ww = win & 3;
  const float* xp = x + (((size_t)(b * 32 + c) * 256 + wh * 64) * 256 + ww * 64);
  for (int idx = tid; idx < 64 * 64; idx += 128)
    sX[(idx >> 6) * 72 + (idx & 63)] = (_Float16)xp[(idx >> 6) * 256 + (idx & 63)];
  __syncthreads();
  gemm_f16(sX, 72, mats + FROW_OFF, 64, sO, 97, 64, 96, 64, wave, lane);
  __syncthreads();
  float* yb = Y + (size_t)win * 33u * 4096u + c * 128;
  const int h = tid >> 1, p = tid & 1;
  for (int k = 0; k < 33; ++k)                      // 512B coalesced chunks
    yb[(size_t)k * 4096 + tid] = sO[h * 97 + 2 * k + p];
}

// ---- K2: per (win,k): colDFT -> xW1~ -> GELU -> xW2~ -> colIDFT (all WMMA)-
__global__ __launch_bounds__(128) void fsel_k2_mix(float* __restrict__ Y,
                                                   const _Float16* __restrict__ mats) {
  __shared__ _Float16 sA[64 * 136];
  __shared__ float    sO[64 * 72];      // also used as 32 x ld132
  const int tid = threadIdx.x, wave = tid >> 5, lane = tid & 31;
  float* ybase = Y + (size_t)blockIdx.x * 4096u;
  for (int idx = tid; idx < 32 * 128; idx += 128)   // load 16KB chunk, ->f16
    sA[(idx >> 7) * 136 + (idx & 127)] = (_Float16)ybase[idx];
  __syncthreads();
  gemm_f16(sA, 136, mats + FCOL_OFF, 128, sO, 132, 32, 128, 128, wave, lane);
  __syncthreads();
  for (int idx = tid; idx < 64 * 64; idx += 128) {  // -> mixing layout [h'][ReC|ImC]
    int hp = idx >> 6, j = idx & 63;
    sA[hp * 72 + j] = (_Float16)sO[(j & 31) * 132 + 2 * hp + (j >> 5)];
  }
  __syncthreads();
  gemm_f16(sA, 72, mats + W1_OFF, 64, sO, 72, 64, 64, 64, wave, lane);
  __syncthreads();
  for (int idx = tid; idx < 64 * 64; idx += 128) {  // exact GELU, f32
    float v = sO[(idx >> 6) * 72 + (idx & 63)];
    sA[(idx >> 6) * 72 + (idx & 63)] =
        (_Float16)(0.5f * v * (1.0f + erff(v * 0.7071067811865475f)));
  }
  __syncthreads();
  gemm_f16(sA, 72, mats + W2_OFF, 64, sO, 72, 64, 64, 64, wave, lane);
  __syncthreads();
  for (int idx = tid; idx < 32 * 128; idx += 128) { // -> [c][2h'+p], scale 1/64
    int c = idx >> 7, j = idx & 127;
    sA[c * 136 + j] = (_Float16)(sO[(j >> 1) * 72 + (c + 32 * (j & 1))] * 0.015625f);
  }
  __syncthreads();
  gemm_f16(sA, 136, mats + FICOL_OFF, 128, sO, 132, 32, 128, 128, wave, lane);
  __syncthreads();
  for (int idx = tid; idx < 32 * 128; idx += 128)   // in-place writeback (f32)
    ybase[idx] = sO[(idx >> 7) * 132 + (idx & 127)];
}

// ---- K3: per (win,c): Hermitian inverse row transform via WMMA -> output --
__global__ __launch_bounds__(128) void fsel_k3_irowfft(const float* __restrict__ Y,
                                                       const _Float16* __restrict__ mats,
                                                       float* __restrict__ out) {
  __shared__ _Float16 sT[64 * 104];
  __shared__ float    sO[64 * 68];
  const int tid = threadIdx.x, wave = tid >> 5, lane = tid & 31;
  const int win = blockIdx.x >> 5, c = blockIdx.x & 31;
  const int b = win >> 4, wh = (win >> 2) & 3, ww = win & 3;
  for (int idx = tid; idx < 64 * 104; idx += 128) sT[idx] = (_Float16)0.0f;
  __syncthreads();
  const float* yb = Y + (size_t)win * 33u * 4096u + c * 128;
  const int h = tid >> 1, p = tid & 1;
  for (int k = 0; k < 33; ++k)                      // gather bins, scale 1/64
    sT[h * 104 + 2 * k + p] = (_Float16)(yb[(size_t)k * 4096 + tid] * 0.015625f);
  __syncthreads();
  gemm_f16(sT, 104, mats + FIROW_OFF, 96, sO, 68, 64, 64, 96, wave, lane);
  __syncthreads();
  float* op = out + (((size_t)(b * 32 + c) * 256 + wh * 64) * 256 + ww * 64);
  for (int idx = tid; idx < 64 * 64; idx += 128)
    op[(idx >> 6) * 256 + (idx & 63)] = sO[(idx >> 6) * 68 + (idx & 63)];
}

extern "C" void kernel_launch(void* const* d_in, const int* in_sizes, int n_in,
                              void* d_out, int out_size, void* d_ws, size_t ws_size,
                              hipStream_t stream) {
  (void)in_sizes; (void)n_in; (void)out_size; (void)ws_size;
  const float* x   = (const float*)d_in[0];
  const float* w1r = (const float*)d_in[1];
  const float* w1i = (const float*)d_in[2];
  const float* w2r = (const float*)d_in[3];
  const float* w2i = (const float*)d_in[4];
  float*     Yscr  = (float*)d_ws;
  _Float16*  mats  = (_Float16*)((char*)d_ws + Y_BYTES);
  float*     out   = (float*)d_out;

  fsel_k0_build<<<64, 256, 0, stream>>>(w1r, w1i, w2r, w2i, mats);
  fsel_k1_rowfft<<<NWIN * 32, 128, 0, stream>>>(x, Yscr, mats);
  fsel_k2_mix<<<NWIN * NK, 128, 0, stream>>>(Yscr, mats);
  fsel_k3_irowfft<<<NWIN * 32, 128, 0, stream>>>(Yscr, mats, out);
}
